// SimpleLSTM_56667798503960
// MI455X (gfx1250) — matrix operand order repaired and checked by reading, not compiled, not run on hardware
//
#include <hip/hip_runtime.h>

// ---------------------------------------------------------------------------
// SimpleLSTM for MI455X (gfx1250): 2-layer LSTM + MLP head.
// Recurrence: per-step GEMM [16 x K] x [K x 512] in bf16 WMMA, weights
// resident in LDS (staged via Tensor Data Mover), cell state in VGPRs,
// double-buffered A operand (1 barrier/step), hardware transcendentals.
// ---------------------------------------------------------------------------

#define B_SZ 64
#define T_SZ 2048
#define IN_SZ 9
#define H_SZ 128

typedef __attribute__((ext_vector_type(16))) __bf16 v16bf;
typedef __attribute__((ext_vector_type(8)))  float  v8f;
typedef __attribute__((ext_vector_type(4)))  unsigned int v4u;
typedef __attribute__((ext_vector_type(8)))  int v8i;
typedef __attribute__((ext_vector_type(4)))  int v4i;

union FragU { uint4 q[2]; v16bf v; };

__device__ __forceinline__ unsigned short f2bf(float f) {
  unsigned u = __builtin_bit_cast(unsigned, f);
  return (unsigned short)((u + 0x7FFFu + ((u >> 16) & 1u)) >> 16);
}

// Branch-free transcendentals on the recurrence critical path:
// v_exp_f32 (exp2), v_rcp_f32, and gfx1250 hardware v_tanh_f32 if exposed.
__device__ __forceinline__ float sigm(float x) {
  return __builtin_amdgcn_rcpf(1.0f + __builtin_amdgcn_exp2f(x * -1.44269504f));
}
#if __has_builtin(__builtin_amdgcn_tanhf)
__device__ __forceinline__ float fast_tanh(float x) {
  return __builtin_amdgcn_tanhf(x);
}
#else
__device__ __forceinline__ float fast_tanh(float x) {
  float e = __builtin_amdgcn_exp2f(x * 2.88539008f);   // exp(2x)
  return 1.0f - 2.0f * __builtin_amdgcn_rcpf(e + 1.0f);
}
#endif

// ---------------------------------------------------------------------------
// Weight packing: combined B matrix B[k][n] = [W_hh^T ; W_ih^T ; 0-pad],
// pre-swizzled into WMMA B-fragment order:
//   fragment(tile,chunk) = 1KB blob, dword index = half*128 + lane*4 + q,
//   vgpr v = half*4+q holds K = chunk*32 + (lane>=16 ? 8:0) + 2v + (v>=4 ? 8:0) (+0/+1 lo/hi)
//   column n = tile*16 + lane%16
// ---------------------------------------------------------------------------
__device__ __forceinline__ float fetchB(const float* Whh, int Khh,
                                        const float* Wih, int Kih,
                                        int n, int k) {
  if (k < Khh) return Whh[n * Khh + k];
  int kk = k - Khh;
  if (kk < Kih) return Wih[n * Kih + kk];
  return 0.0f;
}

__global__ void pack_frags(const float* __restrict__ Whh, int Khh,
                           const float* __restrict__ Wih, int Kih,
                           int Nvalid, int ntiles, int nchunks,
                           unsigned int* __restrict__ dst) {
  int total = ntiles * nchunks * 256;
  int idx = blockIdx.x * blockDim.x + threadIdx.x;
  if (idx >= total) return;
  int dw = idx & 255, frag = idx >> 8;
  int chunk = frag % nchunks, tile = frag / nchunks;
  int half = dw >> 7, rem = dw & 127, lane = rem >> 2, q = rem & 3;
  int v = half * 4 + q;
  int n = tile * 16 + (lane & 15);
  int k0 = chunk * 32 + ((lane >= 16) ? 8 : 0) + v * 2 + ((v >= 4) ? 8 : 0);
  float f0 = 0.0f, f1 = 0.0f;
  if (n < Nvalid) {
    f0 = fetchB(Whh, Khh, Wih, Kih, n, k0);
    f1 = fetchB(Whh, Khh, Wih, Kih, n, k0 + 1);
  }
  dst[idx] = (unsigned)f2bf(f0) | ((unsigned)f2bf(f1) << 16);
}

__global__ void pack_bias(const float* __restrict__ bi0, const float* __restrict__ bh0,
                          const float* __restrict__ bi1, const float* __restrict__ bh1,
                          const float* __restrict__ fc1b, const float* __restrict__ fc2w,
                          float* __restrict__ bias0, float* __restrict__ bias1,
                          float* __restrict__ biash, float* __restrict__ w2ext) {
  int n = blockIdx.x * blockDim.x + threadIdx.x;
  if (n < 512) { bias0[n] = bi0[n] + bh0[n]; bias1[n] = bi1[n] + bh1[n]; }
  if (n < 32)  { biash[n] = (n < 20) ? fc1b[n] : 0.0f;
                 w2ext[n] = (n < 20) ? fc2w[n] : 0.0f; }
}

// ---------------------------------------------------------------------------
// Weight staging into LDS: TDM (tensor_load_to_lds) if the builtin exists,
// cooperative b128 copy otherwise. Each wave DMAs its 1/8 slice.
// ---------------------------------------------------------------------------
#if __has_builtin(__builtin_amdgcn_tensor_load_to_lds)
#define USE_TDM 1
#if __has_include(<hip/amd_detail/amd_gfx1250_TDM.h>)
#define TDM_6ARG 1
#else
#define TDM_6ARG 0
#endif
#endif

__device__ __forceinline__ void stage_weights(char* Wlds,
                                              const unsigned int* wfrag,
                                              int WBYTES, int tid) {
#if defined(USE_TDM)
  {
    int wv = __builtin_amdgcn_readfirstlane((int)((unsigned)tid >> 5));
    unsigned slice  = (unsigned)WBYTES >> 3;        // bytes per wave (1KB mult.)
    unsigned rows   = slice >> 10;                  // rows of 512 bf16 (1KB)
    unsigned ldsoff = (unsigned)wv * slice;
    unsigned long long ga = (unsigned long long)wfrag + (unsigned long long)wv * slice;
    // D# group0: count=1 | lds_addr | global_addr(57b) | type=2
    v4u g0 = { 1u, ldsoff, (unsigned)ga,
               (unsigned)((ga >> 32) & 0x1FFFFFFull) | (2u << 30) };
    // D# group1: data_size=2B; tensor_dim0=512, tensor_dim1=rows;
    //            tile_dim0=512, tile_dim1=rows; dim0 stride=512.
    v8i g1 = { (int)(1u << 16),      // data_size=1 (2 bytes)
               (int)(512u << 16),    // tensor_dim0[15:0] @ bits63:48
               (int)(rows << 16),    // tensor_dim1[15:0] @ bits95:80
               (int)(512u << 16),    // tile_dim0        @ bits127:112
               (int)rows,            // tile_dim1
               512, 0, 0 };          // tensor_dim0_stride
    v4i z4 = { 0, 0, 0, 0 };
#if TDM_6ARG
    v8i z8 = { 0, 0, 0, 0, 0, 0, 0, 0 };
    __builtin_amdgcn_tensor_load_to_lds(g0, g1, z4, z4, z8, 0);
#else
    __builtin_amdgcn_tensor_load_to_lds(g0, g1, z4, z4, 0);
#endif
#if __has_builtin(__builtin_amdgcn_s_wait_tensorcnt)
    __builtin_amdgcn_s_wait_tensorcnt((short)0);
#else
    asm volatile("s_wait_tensorcnt 0x0" ::: "memory");
#endif
  }
#else
  for (int i = tid; i < (WBYTES >> 4); i += 256)
    ((uint4*)Wlds)[i] = ((const uint4*)wfrag)[i];
#endif
}

// ---------------------------------------------------------------------------
// Recurrent layer. Grid = 4 WGs (16 batch rows each), 256 threads = 8 waves.
// Wave w owns gates i,f,g,o for h-slice [16w,16w+16): n-tiles {g*8+w}.
// A (LDS, bf16, double-buffered): [16][KPAD] = [h_{t-1} | x_t | pad].
// KPAD=160/NCHUNK=5 for layer0 (x=f32[9]); KPAD=256/NCHUNK=8 for layer1.
// One barrier per step: step t reads buf[t&1], writes buf[(t+1)&1].
// ---------------------------------------------------------------------------
template <int KPAD, int NCHUNK, int XMODE>
__global__ __launch_bounds__(256)
void lstm_kernel(const unsigned int* __restrict__ wfrag,
                 const float* __restrict__ bias,
                 const float* __restrict__ x32,
                 const unsigned short* __restrict__ hin,
                 unsigned short* __restrict__ hout) {
  constexpr int WBYTES = 32 * NCHUNK * 1024;
  constexpr int AELEMS = 16 * KPAD;
  extern __shared__ char smem[];
  char* Wlds = smem;
  unsigned short* A0 = (unsigned short*)(smem + WBYTES);
  unsigned short* A1 = A0 + AELEMS;

  const int tid  = threadIdx.x;
  const int lane = tid & 31;
  const int wv   = tid >> 5;
  const int b0   = blockIdx.x * 16;

  stage_weights(Wlds, wfrag, WBYTES, tid);   // TDM + s_wait_tensorcnt per wave

  for (int i = tid; i < 2 * AELEMS; i += 256) A0[i] = 0;  // h=0, pads=0
  // x-part for t=0 into buffer 0
  if constexpr (XMODE == 0) {
    if (tid < 16 * IN_SZ) {
      int m = tid / IN_SZ, j = tid % IN_SZ;
      A0[m * KPAD + H_SZ + j] = f2bf(x32[((size_t)(b0 + m) * T_SZ) * IN_SZ + j]);
    }
  } else {
    int m = tid >> 4, h0 = (tid & 15) * 8;
    *(uint4*)(A0 + m * KPAD + H_SZ + h0) =
        *(const uint4*)(hin + ((size_t)(b0 + m) * T_SZ) * H_SZ + h0);
  }
  __syncthreads();

  float bgate[4];
#pragma unroll
  for (int g = 0; g < 4; ++g) bgate[g] = bias[g * 128 + wv * 16 + (lane & 15)];

  v8f cst;
#pragma unroll
  for (int r = 0; r < 8; ++r) cst[r] = 0.0f;

  const int xm = (XMODE == 0) ? (tid / IN_SZ) : (tid >> 4);
  const int xj = (XMODE == 0) ? (tid % IN_SZ) : ((tid & 15) * 8);

  for (int t = 0; t < T_SZ; ++t) {
    unsigned short* cur = (t & 1) ? A1 : A0;
    unsigned short* nxt = (t & 1) ? A0 : A1;

    // Prefetch next timestep's x-part early: latency hides behind WMMA.
    float xf = 0.0f;
    uint4 xq = {0u, 0u, 0u, 0u};
    const bool havex = (t + 1 < T_SZ);
    if (havex) {
      if constexpr (XMODE == 0) {
        if (tid < 16 * IN_SZ)
          xf = x32[((size_t)(b0 + xm) * T_SZ + (t + 1)) * IN_SZ + xj];
      } else {
        xq = *(const uint4*)(hin + ((size_t)(b0 + xm) * T_SZ + (t + 1)) * H_SZ + xj);
      }
    }

    v8f acc[4];
#pragma unroll
    for (int g = 0; g < 4; ++g)
#pragma unroll
      for (int r = 0; r < 8; ++r) acc[g][r] = bgate[g];

#pragma unroll
    for (int c = 0; c < NCHUNK; ++c) {
      // A fragment: row m=lane%16; vgprs 0-3 hold K pairs from
      // chunk*32 + (lane>=16?8:0); vgprs 4-7 from +16. Two ds_load_b128.
      FragU a;
      const char* ab = (const char*)cur + (lane & 15) * (KPAD * 2)
                       + (c * 32 + ((lane >> 4) << 3)) * 2;
      a.q[0] = *(const uint4*)ab;
      a.q[1] = *(const uint4*)(ab + 32);
#pragma unroll
      for (int g = 0; g < 4; ++g) {
        FragU b;
        const char* bb = Wlds + (((g * 8 + wv) * NCHUNK + c) << 10) + lane * 16;
        b.q[0] = *(const uint4*)bb;
        b.q[1] = *(const uint4*)(bb + 512);
        acc[g] = __builtin_amdgcn_wmma_f32_16x16x32_bf16(
            false, a.v, false, b.v, (short)0, acc[g], false, false);
      }
    }

    // Gate update entirely in-wave (c in registers); writes go to the OTHER
    // A buffer, so no barrier is needed between the WMMA reads and this.
#pragma unroll
    for (int r = 0; r < 8; ++r) {
      float iv = sigm(acc[0][r]);
      float fv = sigm(acc[1][r]);
      float gv = fast_tanh(acc[2][r]);
      float ov = sigm(acc[3][r]);
      float cc = fv * cst[r] + iv * gv;
      cst[r] = cc;
      float hv = ov * fast_tanh(cc);
      int m  = r + ((lane >> 4) << 3);
      int hc = wv * 16 + (lane & 15);
      unsigned short hb = f2bf(hv);
      nxt[m * KPAD + hc] = hb;                                  // next-step A
      hout[((size_t)(b0 + m) * T_SZ + t) * H_SZ + hc] = hb;     // layer output
    }
    if (havex) {
      if constexpr (XMODE == 0) {
        if (tid < 16 * IN_SZ) nxt[xm * KPAD + H_SZ + xj] = f2bf(xf);
      } else {
        *(uint4*)(nxt + xm * KPAD + H_SZ + xj) = xq;
      }
    }
    __syncthreads();   // single barrier per step (publishes nxt)
  }
}

// ---------------------------------------------------------------------------
// Head: rows = B*T, out = tanh(h @ fc1^T + b1) @ fc2^T + b2.
// One 16-row tile per wave; N padded 20->32 (2 n-tiles), K=128 (4 chunks).
// Final 20->1 dot via cross-lane shfl_xor reduction over 16 lanes.
// ---------------------------------------------------------------------------
__global__ __launch_bounds__(256)
void head_kernel(const unsigned short* __restrict__ h1,
                 const unsigned int* __restrict__ wfrag,
                 const float* __restrict__ biash,
                 const float* __restrict__ w2ext,
                 const float* __restrict__ fc2b,
                 float* __restrict__ out) {
  int tid = threadIdx.x, lane = tid & 31, wv = tid >> 5;
  long tile = (long)blockIdx.x * 8 + wv;
  long rowbase = tile * 16;
  int m16 = lane & 15, sel = lane >> 4;

  v8f acc0, acc1;
  float bb0 = biash[m16], bb1 = biash[16 + m16];
#pragma unroll
  for (int r = 0; r < 8; ++r) { acc0[r] = bb0; acc1[r] = bb1; }

  const char* abase = (const char*)h1 + (rowbase + m16) * 256;  // 128 bf16/row
#pragma unroll
  for (int c = 0; c < 4; ++c) {
    FragU a;
    const char* ab = abase + c * 64 + sel * 16;
    a.q[0] = *(const uint4*)ab;
    a.q[1] = *(const uint4*)(ab + 32);
    FragU b0f, b1f;
    const char* wb0 = (const char*)wfrag + (c << 10) + lane * 16;
    const char* wb1 = (const char*)wfrag + ((4 + c) << 10) + lane * 16;
    b0f.q[0] = *(const uint4*)wb0; b0f.q[1] = *(const uint4*)(wb0 + 512);
    b1f.q[0] = *(const uint4*)wb1; b1f.q[1] = *(const uint4*)(wb1 + 512);
    acc0 = __builtin_amdgcn_wmma_f32_16x16x32_bf16(
        false, a.v, false, b0f.v, (short)0, acc0, false, false);
    acc1 = __builtin_amdgcn_wmma_f32_16x16x32_bf16(
        false, a.v, false, b1f.v, (short)0, acc1, false, false);
  }

  float w2a = w2ext[m16], w2b = w2ext[16 + m16], c2 = fc2b[0];
#pragma unroll
  for (int r = 0; r < 8; ++r) {
    float p = fast_tanh(acc0[r]) * w2a + fast_tanh(acc1[r]) * w2b;
    p += __shfl_xor(p, 1);
    p += __shfl_xor(p, 2);
    p += __shfl_xor(p, 4);
    p += __shfl_xor(p, 8);
    if (m16 == 0) out[rowbase + r + sel * 8] = p + c2;
  }
}

// ---------------------------------------------------------------------------
extern "C" void kernel_launch(void* const* d_in, const int* in_sizes, int n_in,
                              void* d_out, int out_size, void* d_ws, size_t ws_size,
                              hipStream_t stream) {
  (void)in_sizes; (void)n_in; (void)out_size; (void)ws_size;
  const float* x     = (const float*)d_in[0];
  const float* W_ih0 = (const float*)d_in[1];
  const float* W_hh0 = (const float*)d_in[2];
  const float* b_ih0 = (const float*)d_in[3];
  const float* b_hh0 = (const float*)d_in[4];
  const float* W_ih1 = (const float*)d_in[5];
  const float* W_hh1 = (const float*)d_in[6];
  const float* b_ih1 = (const float*)d_in[7];
  const float* b_hh1 = (const float*)d_in[8];
  const float* fc1_w = (const float*)d_in[9];
  const float* fc1_b = (const float*)d_in[10];
  const float* fc2_w = (const float*)d_in[11];
  const float* fc2_b = (const float*)d_in[12];

  char* ws = (char*)d_ws;
  size_t off = 0;
  auto take = [&](size_t bytes) {
    char* p = ws + off;
    off = (off + bytes + 255) & ~(size_t)255;
    return p;
  };
  unsigned int* frag0 = (unsigned int*)take(163840);   // L0: 32 tiles x 5 chunks
  unsigned int* frag1 = (unsigned int*)take(262144);   // L1: 32 tiles x 8 chunks
  unsigned int* fragH = (unsigned int*)take(8192);     // head: 2 tiles x 4 chunks
  float* bias0 = (float*)take(2048);
  float* bias1 = (float*)take(2048);
  float* biash = (float*)take(128);
  float* w2ext = (float*)take(128);
  unsigned short* h0s = (unsigned short*)take((size_t)B_SZ * T_SZ * H_SZ * 2);
  unsigned short* h1s = (unsigned short*)take((size_t)B_SZ * T_SZ * H_SZ * 2);

  pack_frags<<<160, 256, 0, stream>>>(W_hh0, 128, W_ih0, 9,   512, 32, 5, frag0);
  pack_frags<<<256, 256, 0, stream>>>(W_hh1, 128, W_ih1, 128, 512, 32, 8, frag1);
  pack_frags<<<8,   256, 0, stream>>>(nullptr, 0,  fc1_w, 128, 20,  2, 4, fragH);
  pack_bias <<<2,   256, 0, stream>>>(b_ih0, b_hh0, b_ih1, b_hh1, fc1_b, fc2_w,
                                      bias0, bias1, biash, w2ext);

  lstm_kernel<160, 5, 0><<<4, 256, 163840 + 2 * 16 * 160 * 2, stream>>>(
      frag0, bias0, x, nullptr, h0s);
  lstm_kernel<256, 8, 1><<<4, 256, 262144 + 2 * 16 * 256 * 2, stream>>>(
      frag1, bias1, nullptr, h0s, h1s);
  head_kernel<<<(B_SZ * T_SZ / 16) / 8, 256, 0, stream>>>(
      h1s, fragH, biash, w2ext, fc2_b, (float*)d_out);
}
